// LoraInjectedConv2d_75746043232771
// MI455X (gfx1250) — matrix operand order repaired and checked by reading, not compiled, not run on hardware
//
#include <hip/hip_runtime.h>
#include <stdint.h>

// ---------------------------------------------------------------------------
// LoRA-injected 3x3 conv, MI455X (gfx1250, wave32, WMMA).
//
//  out[b,co,h,w] = conv3x3(x, conv_w)[b,co,h,w] + conv_b[co]
//                + SCALE * act(b) * sum_r up_w[safe(b),co,r] * down[b,r,h,w]
//  down[b,r,h,w] = conv3x3(x[b], down_w[safe(b)])[r,h,w]
//
// Base conv as implicit GEMM: M=COUT=320, N=B*H*W=65536, K=CIN*9=2880,
// computed with V_WMMA_F32_16X16X32_BF16 using a 2-term bf16 split of the
// fp32 operands (3 accumulating WMMAs: hi*hi + hi*lo + lo*hi) -> ~fp32
// accuracy at bf16 matrix rate, f32 accumulation.
//
// Round 3: keep double-buffered LDS (one barrier per K-chunk), but fill the
// next buffer AFTER the WMMA block with short-lived staging registers.
// (Round 2's staging-across-WMMA spilled bv[16] to scratch: scratch_load_b128
// in the hot loop = ~7.5 GB of scratch traffic. Multi-wave overlap hides the
// global-load latency instead.)
// ---------------------------------------------------------------------------

#define B_      16
#define CIN_    320
#define COUT_   320
#define H_      64
#define W_      64
#define HW_     4096          // 64*64
#define R_      4
#define NLORA_  50
#define LSTRIDE_ 4
#define KTOT_   2880          // CIN*9
#define NKCH    (KTOT_ / 32)  // 90 K-chunks
#define SCALE_  1.0f

#define BM 64                 // block tile M (COUT)
#define BN 128                // block tile N (B*H*W)
#define BK 32                 // K chunk = one bf16 WMMA depth
#define KSTR 40               // LDS K stride in bf16 elems (32 + 8 pad, 80B: keeps 16B align)

typedef __bf16 bf16_t;
typedef bf16_t bf16x8  __attribute__((ext_vector_type(8)));
typedef bf16_t bf16x16 __attribute__((ext_vector_type(16)));
typedef float  f32x4   __attribute__((ext_vector_type(4)));
typedef float  f32x8   __attribute__((ext_vector_type(8)));

static __device__ __forceinline__ void split_bf(float v, bf16_t& hi, bf16_t& lo) {
    bf16_t h = (bf16_t)v;
    hi = h;
    lo = (bf16_t)(v - (float)h);
}

// Fragment gather: two contiguous 16B chunks from a K-major LDS row.
static __device__ __forceinline__ bf16x16 ld_frag(const bf16_t* row, int o0, int o1) {
    bf16x8 a = *(const bf16x8*)(row + o0);
    bf16x8 b = *(const bf16x8*)(row + o1);
    return __builtin_shufflevector(a, b, 0,1,2,3,4,5,6,7,8,9,10,11,12,13,14,15);
}

static __device__ __forceinline__ f32x8 wmma_bf16(bf16x16 a, bf16x16 b, f32x8 c) {
    // (neg_a, A, neg_b, B, c_mod, C, reuse_a, reuse_b)
    return __builtin_amdgcn_wmma_f32_16x16x32_bf16(false, a, false, b, (short)0, c,
                                                   false, false);
}

static __device__ __forceinline__ void lora_idx(int id, int& safe, float& act) {
    int idx = (id >= 0) ? (id / LSTRIDE_) : -(((-id) + LSTRIDE_ - 1) / LSTRIDE_); // floor div
    act  = (idx >= 0) ? 1.0f : 0.0f;
    safe = idx < 0 ? 0 : (idx > NLORA_ - 1 ? NLORA_ - 1 : idx);
}

// ---------------------------------------------------------------------------
// Kernel 1: per-sample LoRA down conv (r=4) -> d_ws.  VALU direct conv:
// r=4 << 16-row WMMA granularity and only ~1.25% of total FLOPs.
// One block = (b, 16x16 pixel tile); 18x18 x-patch staged per input channel.
// ---------------------------------------------------------------------------
__global__ __launch_bounds__(256)
void lora_down_kernel(const float* __restrict__ x,
                      const float* __restrict__ down_w,
                      const int*   __restrict__ lora_id,
                      float*       __restrict__ down)
{
    const int t  = threadIdx.x;
    const int tx = t & 15, ty = t >> 4;
    const int b    = blockIdx.x >> 4;
    const int tile = blockIdx.x & 15;
    const int th = (tile >> 2) << 4;
    const int tw = (tile & 3) << 4;

    int safe; float act_unused;
    lora_idx(lora_id[b], safe, act_unused);

    __shared__ float patch[18 * 18];
    __shared__ float wl[R_ * 9];

    float acc[R_] = {0.f, 0.f, 0.f, 0.f};
    const float* xb = x + ((size_t)b * CIN_ << 12);

    for (int ci = 0; ci < CIN_; ++ci) {
        __syncthreads();
        // stage 18x18 zero-padded input patch
        for (int e = t; e < 324; e += 256) {
            int ph = e / 18, pw = e - ph * 18;
            int hh = th + ph - 1, ww = tw + pw - 1;
            patch[e] = ((unsigned)hh < 64u && (unsigned)ww < 64u)
                           ? xb[(ci << 12) + (hh << 6) + ww] : 0.0f;
        }
        // stage 4x3x3 weights for this ci
        if (t < R_ * 9) {
            int r = t / 9, tap = t - r * 9;
            wl[t] = down_w[(((size_t)safe * R_ + r) * CIN_ + ci) * 9 + tap];
        }
        __syncthreads();

        float p[9];
#pragma unroll
        for (int kh = 0; kh < 3; ++kh)
#pragma unroll
            for (int kw = 0; kw < 3; ++kw)
                p[kh * 3 + kw] = patch[(ty + kh) * 18 + tx + kw];
#pragma unroll
        for (int r = 0; r < R_; ++r)
#pragma unroll
            for (int tap = 0; tap < 9; ++tap)
                acc[r] = fmaf(p[tap], wl[r * 9 + tap], acc[r]);
    }

#pragma unroll
    for (int r = 0; r < R_; ++r)
        down[(((size_t)b * R_ + r) << 12) + ((th + ty) << 6) + tw + tx] = acc[r];
}

// ---------------------------------------------------------------------------
// Kernel 2: base conv as implicit GEMM via bf16-split WMMA, with fused
// epilogue: + bias + SCALE*act*  (up_w[safe] . down).
// Block: 256 threads = 8 wave32s, tile BMxBN=64x128, waves 2(M)x4(N),
// each wave -> 32x32 outputs = 4 f32x8 accumulators, 12 WMMAs per K-chunk.
// Double-buffered LDS: one barrier per K-chunk; buffer fill after WMMAs.
// ---------------------------------------------------------------------------
__global__ __launch_bounds__(256)
void lora_conv_main_kernel(const float* __restrict__ x,
                           const float* __restrict__ conv_w,
                           const float* __restrict__ conv_b,
                           const float* __restrict__ up_w,
                           const int*   __restrict__ lora_id,
                           const float* __restrict__ down,
                           float*       __restrict__ out)
{
    const int t    = threadIdx.x;
    const int m0   = blockIdx.y * BM;           // COUT tile base (0..256)
    const int n0   = blockIdx.x * BN;           // flat (b,h,w) tile base
    const int b    = n0 >> 12;                  // BN=128 divides 4096 -> b fixed
    const int hwb  = n0 & (HW_ - 1);

    const int lane = t & 31;
    const int half = lane >> 4;                 // 0: lanes 0-15, 1: lanes 16-31
    const int lr   = lane & 15;
    const int wave = t >> 5;
    const int wm   = (wave >> 2) * 32;          // wave M offset in tile
    const int wn   = (wave & 3) * 32;           // wave N offset in tile

    // Double-buffered LDS tiles, K-major, hi/lo bf16 split. 80B rows: 16B align.
    __shared__ __align__(16) bf16_t sAh[2][BM * KSTR];
    __shared__ __align__(16) bf16_t sAl[2][BM * KSTR];
    __shared__ __align__(16) bf16_t sBh[2][BN * KSTR];
    __shared__ __align__(16) bf16_t sBl[2][BN * KSTR];

    f32x8 acc[2][2] = {};                       // [mi][ni]

    // A staging map: thread -> (row am, 8 consecutive k) => two float4 loads.
    const int am = t >> 2;
    const int ak = (t & 3) * 8;
    const float* aprow = conv_w + (size_t)(m0 + am) * KTOT_ + ak;

    // Fill buffer `buf` with K-chunk `kt`: global load -> bf16 split -> LDS.
    // Staging values are short-lived (never cross the WMMA region -> no spill).
    auto stage_fill = [&](int buf, int kt) {
        const int k0 = kt * BK;
        const float* ap = aprow + k0;
        f32x4 a0 = *(const f32x4*)(ap);
        f32x4 a1 = *(const f32x4*)(ap + 4);
        bf16_t* dh = &sAh[buf][am * KSTR + ak];
        bf16_t* dl = &sAl[buf][am * KSTR + ak];
#pragma unroll
        for (int i = 0; i < 4; ++i) { split_bf(a0[i], dh[i],     dl[i]);     }
#pragma unroll
        for (int i = 0; i < 4; ++i) { split_bf(a1[i], dh[4 + i], dl[4 + i]); }
        // B = im2col(x): element e = i*256 + t, nn = e&127 (lane-consecutive,
        // coalesced), kk = e>>7.
#pragma unroll 4
        for (int i = 0; i < 16; ++i) {
            int e  = i * 256 + t;
            int nn = e & (BN - 1);
            int kk = e >> 7;
            int kg = k0 + kk;
            int ci = kg / 9;
            int rm = kg - ci * 9;
            int kh = rm / 3;
            int kw = rm - kh * 3;
            int hw = hwb + nn;
            int hp = (hw >> 6) + kh - 1;
            int wp = (hw & 63) + kw - 1;
            float v = 0.0f;
            if ((unsigned)hp < 64u && (unsigned)wp < 64u)
                v = x[(((size_t)b * CIN_ + ci) << 12) + (hp << 6) + wp];
            split_bf(v, sBh[buf][nn * KSTR + kk], sBl[buf][nn * KSTR + kk]);
        }
    };

    // ---- prologue: chunk 0 into buffer 0 ----
    stage_fill(0, 0);
    __syncthreads();

    // ---- main loop: 1 barrier per K-chunk ----
    for (int kt = 0; kt < NKCH; ++kt) {
        const int cur = kt & 1;

        // fragments (two ds_load_b128 each, per ISA 16-bit layouts) - ready now
        bf16x16 Ah[2], Al[2], Bh[2], Bl[2];
#pragma unroll
        for (int mi = 0; mi < 2; ++mi) {
            const bf16_t* rh = &sAh[cur][(wm + mi * 16 + lr) * KSTR];
            const bf16_t* rl = &sAl[cur][(wm + mi * 16 + lr) * KSTR];
            int o0 = half * 8;              // A: lanes<16 K0-7/K16-23, >=16 K8-15/K24-31
            Ah[mi] = ld_frag(rh, o0, o0 + 16);
            Al[mi] = ld_frag(rl, o0, o0 + 16);
        }
#pragma unroll
        for (int ni = 0; ni < 2; ++ni) {
            const bf16_t* rh = &sBh[cur][(wn + ni * 16 + lr) * KSTR];
            const bf16_t* rl = &sBl[cur][(wn + ni * 16 + lr) * KSTR];
            int o0 = half * 8;              // B: lanes<16 K0-7/K16-23, >=16 K8-15/K24-31
            Bh[ni] = ld_frag(rh, o0, o0 + 16);
            Bl[ni] = ld_frag(rl, o0, o0 + 16);
        }

        // 12 WMMAs: hi*hi + hi*lo + lo*hi (fp32-class accuracy)
#pragma unroll
        for (int mi = 0; mi < 2; ++mi)
#pragma unroll
            for (int ni = 0; ni < 2; ++ni) {
                acc[mi][ni] = wmma_bf16(Ah[mi], Bh[ni], acc[mi][ni]);
                acc[mi][ni] = wmma_bf16(Ah[mi], Bl[ni], acc[mi][ni]);
                acc[mi][ni] = wmma_bf16(Al[mi], Bh[ni], acc[mi][ni]);
            }

        // fill the other buffer for chunk k+1 (readers of it synced last barrier)
        if (kt + 1 < NKCH) stage_fill(cur ^ 1, kt + 1);

        __syncthreads();
    }

    // ---- fused epilogue: bias + LoRA up (1x1) + mask, then store.
    int safe; float act;
    lora_idx(lora_id[b], safe, act);

#pragma unroll
    for (int ni = 0; ni < 2; ++ni) {
        const int hw = hwb + wn + ni * 16 + lr;     // C/D layout: N = lane&15
        const float d0 = down[(((size_t)b * R_ + 0) << 12) + hw];
        const float d1 = down[(((size_t)b * R_ + 1) << 12) + hw];
        const float d2 = down[(((size_t)b * R_ + 2) << 12) + hw];
        const float d3 = down[(((size_t)b * R_ + 3) << 12) + hw];
#pragma unroll
        for (int mi = 0; mi < 2; ++mi) {
            f32x8 c = acc[mi][ni];
            const int cobase = m0 + wm + mi * 16 + half * 8;  // C/D: M = vgpr + half*8
#pragma unroll
            for (int g = 0; g < 8; ++g) {
                const int co = cobase + g;
                const float* uw = up_w + (((size_t)safe * COUT_ + co) << 2);
                float up = uw[0] * d0 + uw[1] * d1 + uw[2] * d2 + uw[3] * d3;
                out[(((size_t)b * COUT_ + co) << 12) + hw] =
                    c[g] + conv_b[co] + SCALE_ * act * up;
            }
        }
    }
}

// ---------------------------------------------------------------------------
// Launch. Inputs (setup_inputs order): x, conv_w, conv_b, down_w, up_w, lora_id.
// d_ws holds down[B,R,H,W] = 1 MiB of f32 scratch.
// ---------------------------------------------------------------------------
extern "C" void kernel_launch(void* const* d_in, const int* in_sizes, int n_in,
                              void* d_out, int out_size, void* d_ws, size_t ws_size,
                              hipStream_t stream)
{
    (void)in_sizes; (void)n_in; (void)out_size; (void)ws_size;

    const float* x       = (const float*)d_in[0];
    const float* conv_w  = (const float*)d_in[1];
    const float* conv_b  = (const float*)d_in[2];
    const float* down_w  = (const float*)d_in[3];
    const float* up_w    = (const float*)d_in[4];
    const int*   lora_id = (const int*)d_in[5];
    float*       out     = (float*)d_out;
    float*       down    = (float*)d_ws;   // B*R*HW floats = 1 MiB

    // 1) per-sample LoRA down conv (r=4) into workspace
    lora_down_kernel<<<dim3(B_ * 16), dim3(256), 0, stream>>>(x, down_w, lora_id, down);

    // 2) WMMA implicit-GEMM base conv with fused bias + LoRA-up epilogue
    dim3 grid((B_ * HW_) / BN, COUT_ / BM);   // 512 x 5
    lora_conv_main_kernel<<<grid, dim3(256), 0, stream>>>(
        x, conv_w, conv_b, up_w, lora_id, down, out);
}
